// MixtralSparseMoeBlock_39238821216263
// MI455X (gfx1250) — matrix operand order repaired and checked by reading, not compiled, not run on hardware
//
#include <hip/hip_runtime.h>

// MoE block: T=4096 tokens, H=2048, I=7168, E=8, top-2.
// bf16 WMMA path (v_wmma_f32_16x16x32_bf16), gathered grouped-GEMM per expert.
// Double-buffered LDS; A tiles via global_load_async_to_lds_b128 (ASYNCcnt),
// fp32 weight tiles reg-staged + v_cvt_pk_bf16_f32 into LDS.

#define TOK 4096
#define HD  2048
#define ID  7168
#define NE  8

#define BK  32
#define SA  40      // BK + 8 pad (80B row stride: 16B aligned, conflict-free)
#define BM1 128
#define BN1 64
#define BM2 128
#define BN2 128

typedef __attribute__((ext_vector_type(16))) __bf16 v16bf;
typedef __attribute__((ext_vector_type(8)))  __bf16 v8bf;
typedef __attribute__((ext_vector_type(8)))  float  v8f;
typedef __attribute__((ext_vector_type(4)))  float  v4f;

union AF { v16bf v; v8bf h[2]; };

__device__ __forceinline__ unsigned lds_addr32(const void* p) {
  // generic LDS pointer = {shared aperture, lds byte address}; low 32 bits
  // are the allocation-relative address the async-LDS ops expect.
  return (unsigned)(size_t)p;
}

__device__ __forceinline__ void async_load_b128(unsigned lds_off,
                                                const void* gptr) {
  asm volatile("global_load_async_to_lds_b128 %0, %1, off"
               :: "v"(lds_off), "v"((unsigned long long)(size_t)gptr)
               : "memory");
}

__device__ __forceinline__ void wait_async0() {
  asm volatile("s_wait_asynccnt 0x0" ::: "memory");
}

// A fragment 16x32 bf16: lane (hf=lane>>4, m=lane&15) holds
// K = hf*8+0..7 (elems 0..7) and K = 16+hf*8+0..7 (elems 8..15) of row m.
__device__ __forceinline__ v16bf load_a_frag(const __bf16* s, int row_base, int lane) {
  const int hf = lane >> 4, m = lane & 15;
  const __bf16* rp = s + (row_base + m) * SA;
  AF f;
  f.h[0] = *(const v8bf*)(rp + hf * 8);
  f.h[1] = *(const v8bf*)(rp + 16 + hf * 8);
  return f.v;
}

// B fragment 32x16 bf16: lane holds column n=lane&15, K = hf*8+0..7 (elems 0..7) and K = 16+hf*8+0..7 (elems 8..15).
__device__ __forceinline__ v16bf load_b_frag(const __bf16* s, int col_base, int lane) {
  const int hf = lane >> 4, n = lane & 15;
  const __bf16* rp = s + (col_base + n) * SA;
  AF f;
  f.h[0] = *(const v8bf*)(rp + hf * 8);
  f.h[1] = *(const v8bf*)(rp + 16 + hf * 8);
  return f.v;
}

// ---- x fp32 -> bf16 ----------------------------------------------------
__global__ __launch_bounds__(256) void k_cvt_bf16(const float* __restrict__ x,
                                                  __bf16* __restrict__ xb) {
  const size_t i = ((size_t)blockIdx.x * 256 + threadIdx.x) * 8;
  v4f a = *(const v4f*)(x + i);
  v4f b = *(const v4f*)(x + i + 4);
  v8bf o;
  o[0]=(__bf16)a[0]; o[1]=(__bf16)a[1]; o[2]=(__bf16)a[2]; o[3]=(__bf16)a[3];
  o[4]=(__bf16)b[0]; o[5]=(__bf16)b[1]; o[6]=(__bf16)b[2]; o[7]=(__bf16)b[3];
  *(v8bf*)(xb + i) = o;
}

// ---- router: logits, softmax, top-2, build per-expert token lists ------
__global__ __launch_bounds__(256) void k_router(const float* __restrict__ x,
                                                const float* __restrict__ gate_w,
                                                float* __restrict__ logits_out,
                                                int* __restrict__ counts,
                                                int* __restrict__ tokens,
                                                float* __restrict__ wts) {
  const int t = blockIdx.x;
  const int wv = threadIdx.x >> 5;    // expert handled by this wave
  const int lane = threadIdx.x & 31;
  const float* xr = x + (size_t)t * HD;
  const float* gr = gate_w + (size_t)wv * HD;
  float s = 0.f;
  for (int k = lane * 4; k < HD; k += 128) {
    v4f xa = *(const v4f*)(xr + k);
    v4f ga = *(const v4f*)(gr + k);
    s += xa[0]*ga[0] + xa[1]*ga[1] + xa[2]*ga[2] + xa[3]*ga[3];
  }
  #pragma unroll
  for (int off = 16; off > 0; off >>= 1) s += __shfl_xor(s, off, 32);
  __shared__ float lg[NE];
  if (lane == 0) lg[wv] = s;
  __syncthreads();
  if (threadIdx.x == 0) {
    float l[NE], mx = lg[0];
    #pragma unroll
    for (int e = 0; e < NE; ++e) { l[e] = lg[e]; mx = fmaxf(mx, l[e]); }
    float p[NE], sum = 0.f;
    #pragma unroll
    for (int e = 0; e < NE; ++e) { p[e] = __expf(l[e] - mx); sum += p[e]; }
    const float inv = 1.f / sum;
    #pragma unroll
    for (int e = 0; e < NE; ++e) { p[e] *= inv; logits_out[t * NE + e] = l[e]; }
    int i1 = 0;
    #pragma unroll
    for (int e = 1; e < NE; ++e) if (p[e] > p[i1]) i1 = e;
    int i2 = (i1 == 0) ? 1 : 0;
    #pragma unroll
    for (int e = 0; e < NE; ++e) if (e != i1 && p[e] > p[i2]) i2 = e;
    const float wn = 1.f / (p[i1] + p[i2]);
    int p1 = atomicAdd(&counts[i1], 1);
    tokens[i1 * TOK + p1] = t; wts[i1 * TOK + p1] = p[i1] * wn;
    int p2 = atomicAdd(&counts[i2], 1);
    tokens[i2 * TOK + p2] = t; wts[i2 * TOK + p2] = p[i2] * wn;
  }
}

__global__ void k_scan(const int* __restrict__ counts, int* __restrict__ offs) {
  if (threadIdx.x == 0) {
    int a = 0;
    for (int e = 0; e < NE; ++e) { offs[e] = a; a += counts[e]; }
  }
}

// ---- GEMM1: hdn = silu(x@w1^T) * (x@w3^T), gathered rows, bf16 out -----
__global__ __launch_bounds__(256) void k_gemm1(
    const __bf16* __restrict__ xb,
    const float* __restrict__ w1, const float* __restrict__ w3,
    const int* __restrict__ counts, const int* __restrict__ offs,
    const int* __restrict__ tokens,
    __bf16* __restrict__ hdn) {
  const int e = blockIdx.z;
  const int count = counts[e];
  const int mtile = blockIdx.y;
  if (mtile * BM1 >= count) return;
  const int ntile = blockIdx.x;
  const int offe = offs[e];

  __shared__ __bf16 As[2][BM1 * SA];
  __shared__ __bf16 B1s[2][BN1 * SA];
  __shared__ __bf16 B3s[2][BN1 * SA];
  __shared__ int toks[BM1];

  const int tid = threadIdx.x;
  const int lane = tid & 31, wave = tid >> 5;
  const int wm = wave & 3, wn = wave >> 2;
  const int hf = lane >> 4;

  if (tid < BM1) {
    const int pos = mtile * BM1 + tid;
    toks[tid] = tokens[e * TOK + (pos < count ? pos : 0)];
  }
  __syncthreads();   // toks visible before async gather uses them

  const float* w1e = w1 + ((size_t)e * ID + (size_t)ntile * BN1) * HD;
  const float* w3e = w3 + ((size_t)e * ID + (size_t)ntile * BN1) * HD;
  const unsigned as_base = lds_addr32(&As[0][0]);

  // async gather of A tile (bf16, no conversion): 512 chunks of 16B
  auto issue_a = [&](int k0, int buf) {
    #pragma unroll
    for (int it = 0; it < 2; ++it) {
      const int c = tid + it * 256;
      const int row = c >> 2, part = c & 3;
      const unsigned lds = as_base + (unsigned)(buf * (BM1 * SA) + row * SA + part * 8) * 2u;
      async_load_b128(lds, xb + (size_t)toks[row] * HD + k0 + part * 8);
    }
  };

  v4f br1[2], br3[2];
  auto load_b = [&](int k0) {
    #pragma unroll
    for (int it = 0; it < 2; ++it) {
      const int c = tid + it * 256;           // 512 chunks of 4 fp32
      const int row = c >> 3, part = c & 7;
      const size_t so = (size_t)row * HD + k0 + part * 4;
      br1[it] = *(const v4f*)(w1e + so);
      br3[it] = *(const v4f*)(w3e + so);
      if (part == 0 && k0 + 2 * BK < HD) {
        __builtin_prefetch(w1e + so + 2 * BK, 0, 1);
        __builtin_prefetch(w3e + so + 2 * BK, 0, 1);
      }
    }
  };
  auto store_b = [&](int buf) {
    #pragma unroll
    for (int it = 0; it < 2; ++it) {
      const int c = tid + it * 256;
      const int row = c >> 3, part = c & 7;
      __bf16* d1 = &B1s[buf][row * SA + part * 4];
      __bf16* d3 = &B3s[buf][row * SA + part * 4];
      d1[0]=(__bf16)br1[it][0]; d1[1]=(__bf16)br1[it][1]; d1[2]=(__bf16)br1[it][2]; d1[3]=(__bf16)br1[it][3];
      d3[0]=(__bf16)br3[it][0]; d3[1]=(__bf16)br3[it][1]; d3[2]=(__bf16)br3[it][2]; d3[3]=(__bf16)br3[it][3];
    }
  };

  v8f accg[2][2], accu[2][2];
  #pragma unroll
  for (int a = 0; a < 2; ++a)
    #pragma unroll
    for (int b = 0; b < 2; ++b) { accg[a][b] = {}; accu[a][b] = {}; }

  // prologue: stage 0
  issue_a(0, 0);
  load_b(0);
  store_b(0);
  wait_async0();
  __syncthreads();

  int cur = 0;
  for (int k0 = 0; k0 < HD; k0 += BK) {
    const int nxt = cur ^ 1;
    const bool more = (k0 + BK) < HD;
    if (more) { issue_a(k0 + BK, nxt); load_b(k0 + BK); }

    v16bf af[2], b1f[2], b3f[2];
    #pragma unroll
    for (int im = 0; im < 2; ++im) af[im] = load_a_frag(As[cur], wm * 32 + im * 16, lane);
    #pragma unroll
    for (int in = 0; in < 2; ++in) {
      b1f[in] = load_b_frag(B1s[cur], wn * 32 + in * 16, lane);
      b3f[in] = load_b_frag(B3s[cur], wn * 32 + in * 16, lane);
    }
    #pragma unroll
    for (int im = 0; im < 2; ++im)
      #pragma unroll
      for (int in = 0; in < 2; ++in) {
        accg[im][in] = __builtin_amdgcn_wmma_f32_16x16x32_bf16(
            false, af[im], false, b1f[in], (short)0, accg[im][in], false, false);
        accu[im][in] = __builtin_amdgcn_wmma_f32_16x16x32_bf16(
            false, af[im], false, b3f[in], (short)0, accu[im][in], false, false);
      }

    if (more) { store_b(nxt); wait_async0(); }
    __syncthreads();
    cur = nxt;
  }

  // fused SwiGLU epilogue -> hdn (bf16)
  #pragma unroll
  for (int im = 0; im < 2; ++im) {
    const int rl0 = wm * 32 + im * 16 + hf * 8;
    #pragma unroll
    for (int in = 0; in < 2; ++in) {
      const int col = ntile * BN1 + wn * 32 + in * 16 + (lane & 15);
      #pragma unroll
      for (int r = 0; r < 8; ++r) {
        const int pos = mtile * BM1 + rl0 + r;
        if (pos < count) {
          const float g = accg[im][in][r], u = accu[im][in][r];
          const float hv = (g / (1.f + __expf(-g))) * u;
          hdn[(size_t)(offe + pos) * ID + col] = (__bf16)hv;
        }
      }
    }
  }
}

// ---- GEMM2: out[token] += wt * (hdn @ w2^T) ----------------------------
__global__ __launch_bounds__(256) void k_gemm2(
    const __bf16* __restrict__ hdn,
    const float* __restrict__ w2,
    const int* __restrict__ counts, const int* __restrict__ offs,
    const int* __restrict__ tokens, const float* __restrict__ wts,
    float* __restrict__ out) {
  const int e = blockIdx.z;
  const int count = counts[e];
  const int mtile = blockIdx.y;
  if (mtile * BM2 >= count) return;
  const int ntile = blockIdx.x;
  const int offe = offs[e];

  __shared__ __bf16 As[2][BM2 * SA];
  __shared__ __bf16 Bs[2][BN2 * SA];
  __shared__ int toks[BM2];
  __shared__ float wls[BM2];

  const int tid = threadIdx.x;
  const int lane = tid & 31, wave = tid >> 5;
  const int wm = wave & 3, wn = wave >> 2;
  const int hf = lane >> 4;

  if (tid < BM2) {
    const int pos = mtile * BM2 + tid;
    const int pc = pos < count ? pos : 0;
    toks[tid] = tokens[e * TOK + pc];
    wls[tid]  = wts[e * TOK + pc];
  }

  const float* w2e = w2 + ((size_t)e * HD + (size_t)ntile * BN2) * ID;
  const __bf16* ae = hdn + (size_t)offe * ID;
  const unsigned as_base = lds_addr32(&As[0][0]);

  auto issue_a = [&](int k0, int buf) {
    #pragma unroll
    for (int it = 0; it < 2; ++it) {
      const int c = tid + it * 256;
      const int row = c >> 2, part = c & 3;
      const int pos = mtile * BM2 + row;
      const int pc = pos < count ? pos : 0;
      const unsigned lds = as_base + (unsigned)(buf * (BM2 * SA) + row * SA + part * 8) * 2u;
      async_load_b128(lds, ae + (size_t)pc * ID + k0 + part * 8);
    }
  };

  v4f breg[4];
  auto load_b = [&](int k0) {
    #pragma unroll
    for (int it = 0; it < 4; ++it) {
      const int c = tid + it * 256;           // 1024 chunks of 4 fp32
      const int row = c >> 3, part = c & 7;
      const size_t so = (size_t)row * ID + k0 + part * 4;
      breg[it] = *(const v4f*)(w2e + so);
      if (part == 0 && k0 + 2 * BK < ID) __builtin_prefetch(w2e + so + 2 * BK, 0, 1);
    }
  };
  auto store_b = [&](int buf) {
    #pragma unroll
    for (int it = 0; it < 4; ++it) {
      const int c = tid + it * 256;
      const int row = c >> 3, part = c & 7;
      __bf16* d = &Bs[buf][row * SA + part * 4];
      d[0]=(__bf16)breg[it][0]; d[1]=(__bf16)breg[it][1];
      d[2]=(__bf16)breg[it][2]; d[3]=(__bf16)breg[it][3];
    }
  };

  v8f acc[2][4];
  #pragma unroll
  for (int a = 0; a < 2; ++a)
    #pragma unroll
    for (int b = 0; b < 4; ++b) acc[a][b] = {};

  // prologue: stage 0 (also covers toks/wls visibility via the barrier)
  __syncthreads();
  issue_a(0, 0);
  load_b(0);
  store_b(0);
  wait_async0();
  __syncthreads();

  int cur = 0;
  for (int k0 = 0; k0 < ID; k0 += BK) {
    const int nxt = cur ^ 1;
    const bool more = (k0 + BK) < ID;
    if (more) { issue_a(k0 + BK, nxt); load_b(k0 + BK); }

    v16bf af[2], bf[4];
    #pragma unroll
    for (int im = 0; im < 2; ++im) af[im] = load_a_frag(As[cur], wm * 32 + im * 16, lane);
    #pragma unroll
    for (int in = 0; in < 4; ++in) bf[in] = load_b_frag(Bs[cur], wn * 64 + in * 16, lane);
    #pragma unroll
    for (int im = 0; im < 2; ++im)
      #pragma unroll
      for (int in = 0; in < 4; ++in)
        acc[im][in] = __builtin_amdgcn_wmma_f32_16x16x32_bf16(
            false, af[im], false, bf[in], (short)0, acc[im][in], false, false);

    if (more) { store_b(nxt); wait_async0(); }
    __syncthreads();
    cur = nxt;
  }

  // scaled scatter-add into out (exactly 2 adds per element -> deterministic)
  #pragma unroll
  for (int im = 0; im < 2; ++im) {
    const int rl0 = wm * 32 + im * 16 + hf * 8;
    #pragma unroll
    for (int in = 0; in < 4; ++in) {
      const int col = ntile * BN2 + wn * 64 + in * 16 + (lane & 15);
      #pragma unroll
      for (int r = 0; r < 8; ++r) {
        const int rl = rl0 + r;
        if (mtile * BM2 + rl < count) {
          const float v = acc[im][in][r] * wls[rl];
          unsafeAtomicAdd(&out[(size_t)toks[rl] * HD + col], v);
        }
      }
    }
  }
}

extern "C" void kernel_launch(void* const* d_in, const int* in_sizes, int n_in,
                              void* d_out, int out_size, void* d_ws, size_t ws_size,
                              hipStream_t stream) {
  const float* x      = (const float*)d_in[0];
  const float* gate_w = (const float*)d_in[1];
  const float* w1     = (const float*)d_in[2];
  const float* w3     = (const float*)d_in[3];
  const float* w2     = (const float*)d_in[4];
  float* out    = (float*)d_out;                 // [TOK*HD] out_states
  float* logits = out + (size_t)TOK * HD;        // [TOK*NE] router_logits

  char* ws = (char*)d_ws;
  int*    counts = (int*)ws;                               // 8 ints
  int*    offs   = (int*)(ws + 256);                       // 8 ints
  int*    tokens = (int*)(ws + 512);                       // NE*TOK ints
  float*  wts    = (float*)(ws + 512 + (size_t)NE*TOK*4);  // NE*TOK floats
  __bf16* xb     = (__bf16*)(ws + 512 + (size_t)NE*TOK*8); // TOK*HD bf16
  __bf16* hdn    = (__bf16*)(ws + 512 + (size_t)NE*TOK*8 + (size_t)TOK*HD*2); // 2*TOK*ID bf16

  hipMemsetAsync(out, 0, (size_t)TOK * HD * sizeof(float), stream);
  hipMemsetAsync(counts, 0, NE * sizeof(int), stream);

  k_cvt_bf16<<<dim3((TOK * HD) / (256 * 8)), dim3(256), 0, stream>>>(x, xb);
  k_router<<<dim3(TOK), dim3(256), 0, stream>>>(x, gate_w, logits, counts, tokens, wts);
  k_scan<<<dim3(1), dim3(32), 0, stream>>>(counts, offs);
  k_gemm1<<<dim3(ID / BN1, TOK / BM1, NE), dim3(256), 0, stream>>>(
      xb, w1, w3, counts, offs, tokens, hdn);
  k_gemm2<<<dim3(HD / BN2, TOK / BM2, NE), dim3(256), 0, stream>>>(
      hdn, w2, counts, offs, tokens, wts, out);
}
